// PointNetFeatureUpsampling_61469571940541
// MI455X (gfx1250) — matrix operand order repaired and checked by reading, not compiled, not run on hardware
//
#include <hip/hip_runtime.h>

// ---------------------------------------------------------------------------
// PointNet feature upsampling for MI455X (gfx1250, wave32).
// Pipeline: KNN -> IDW interp -> concat -> [GEMM + maskedBN + ReLU] x2
// GEMMs use split-bf16 (hi/lo) V_WMMA_F32_16X16X32_BF16, fp32 accumulate.
// Wave tile = 4 Mtiles x 2 Ntiles -> 24 b128 loads : 24 WMMAs per k-chunk.
// ---------------------------------------------------------------------------

#define B_  16
#define N_  2048
#define S_  512
#define D_  384
#define C0  768   // concat channels / layer0 K and layer0 out
#define C1  384   // layer1 out
#define ROWS (B_ * N_)   // 32768
#define KNN 5
#define NCHUNK 64        // row chunks for deterministic BN reduction

typedef __attribute__((ext_vector_type(16))) __bf16        v16bf;
typedef __attribute__((ext_vector_type(8)))  float         v8f;
typedef __attribute__((ext_vector_type(8)))  unsigned int  v8u;

// ---- bf16 helpers (round-to-nearest-even) ----------------------------------
__device__ __forceinline__ unsigned short f2bf(float f) {
    unsigned int u = __float_as_uint(f);
    unsigned int r = u + 0x7fffu + ((u >> 16) & 1u);
    return (unsigned short)(r >> 16);
}
__device__ __forceinline__ float bf2f(unsigned short h) {
    return __uint_as_float(((unsigned int)h) << 16);
}

// ---- split a float array into bf16 hi/lo pair ------------------------------
__global__ void __launch_bounds__(256)
split_kernel(const float* __restrict__ src, unsigned short* __restrict__ dh,
             unsigned short* __restrict__ dl, int n) {
    for (int e = blockIdx.x * 256 + threadIdx.x; e < n; e += gridDim.x * 256) {
        float v = src[e];
        unsigned short h = f2bf(v);
        dh[e] = h;
        dl[e] = f2bf(v - bf2f(h));
    }
}

// ---- KNN: per thread, scan S refs, keep 5 smallest (stable ties) -----------
__global__ void __launch_bounds__(256)
knn_kernel(const float* __restrict__ xyz1, const float* __restrict__ xyz2,
           const int* __restrict__ elens, int* __restrict__ outIdx,
           float* __restrict__ outW) {
    __shared__ float sx[S_], sy[S_], sz[S_];
    int b = blockIdx.x >> 3;                       // 8 blocks per batch elem
    int n = ((blockIdx.x & 7) << 8) + threadIdx.x;
    const float* p2 = xyz2 + (size_t)b * S_ * 3;
    for (int s = threadIdx.x; s < S_; s += 256) {
        sx[s] = p2[s * 3 + 0];
        sy[s] = p2[s * 3 + 1];
        sz[s] = p2[s * 3 + 2];
    }
    __syncthreads();
    int slen = elens[b];
    const float* p1 = xyz1 + ((size_t)b * N_ + n) * 3;
    float x = p1[0], y = p1[1], z = p1[2];
    float n1 = x * x + y * y + z * z;

    float d0 = 3.0e38f, d1 = 3.0e38f, d2v = 3.0e38f, d3 = 3.0e38f, d4 = 3.0e38f;
    int   i0 = 0, i1 = 0, i2 = 0, i3 = 0, i4 = 0;
    for (int s = 0; s < S_; ++s) {
        float xs = sx[s], ys = sy[s], zs = sz[s];
        float dd = n1 + (xs * xs + ys * ys + zs * zs)
                 - 2.0f * (x * xs + y * ys + z * zs);
        if (s >= slen) dd = 1e10f;                  // same sentinel as reference
        if (dd < d4) {
            d4 = dd; i4 = s;
            if (d4 < d3) { float t = d4; d4 = d3; d3 = t; int ti = i4; i4 = i3; i3 = ti;
                if (d3 < d2v) { t = d3; d3 = d2v; d2v = t; ti = i3; i3 = i2; i2 = ti;
                    if (d2v < d1) { t = d2v; d2v = d1; d1 = t; ti = i2; i2 = i1; i1 = ti;
                        if (d1 < d0) { t = d1; d1 = d0; d0 = t; ti = i1; i1 = i0; i0 = ti; }
                    }
                }
            }
        }
    }
    const float eps = 1.1920929e-07f;               // finfo(float32).eps
    float r0 = 1.0f / (d0 + eps), r1 = 1.0f / (d1 + eps), r2 = 1.0f / (d2v + eps);
    float r3 = 1.0f / (d3 + eps), r4 = 1.0f / (d4 + eps);
    float rs = r0 + r1 + r2 + r3 + r4;
    size_t row = (size_t)b * N_ + n;
    outIdx[row * KNN + 0] = i0; outW[row * KNN + 0] = r0 / rs;
    outIdx[row * KNN + 1] = i1; outW[row * KNN + 1] = r1 / rs;
    outIdx[row * KNN + 2] = i2; outW[row * KNN + 2] = r2 / rs;
    outIdx[row * KNN + 3] = i3; outW[row * KNN + 3] = r3 / rs;
    outIdx[row * KNN + 4] = i4; outW[row * KNN + 4] = r4 / rs;
}

// ---- build X = [points1 | IDW-interp(points2)] as bf16 hi/lo ---------------
__global__ void __launch_bounds__(256)
buildx_kernel(const float* __restrict__ points1, const float* __restrict__ points2,
              const int* __restrict__ kIdx, const float* __restrict__ kW,
              unsigned short* __restrict__ Xh, unsigned short* __restrict__ Xl) {
    size_t row = blockIdx.x;
    int b = (int)(row >> 11);
    int j0 = kIdx[row * KNN + 0], j1 = kIdx[row * KNN + 1], j2 = kIdx[row * KNN + 2];
    int j3 = kIdx[row * KNN + 3], j4 = kIdx[row * KNN + 4];
    float w0 = kW[row * KNN + 0], w1 = kW[row * KNN + 1], w2 = kW[row * KNN + 2];
    float w3 = kW[row * KNN + 3], w4 = kW[row * KNN + 4];
    const float* p1 = points1 + row * D_;
    const float* p2 = points2 + (size_t)b * S_ * D_;
    for (int c = threadIdx.x; c < C0; c += 256) {
        float v;
        if (c < D_) {
            v = p1[c];
        } else {
            int cc = c - D_;
            v = w0 * p2[(size_t)j0 * D_ + cc] + w1 * p2[(size_t)j1 * D_ + cc]
              + w2 * p2[(size_t)j2 * D_ + cc] + w3 * p2[(size_t)j3 * D_ + cc]
              + w4 * p2[(size_t)j4 * D_ + cc];
        }
        unsigned short h = f2bf(v);
        Xh[row * C0 + c] = h;
        Xl[row * C0 + c] = f2bf(v - bf2f(h));
    }
}

// ---- split-bf16 GEMM: Y[m,d] = sum_k X[m,k] * W[d,k] -----------------------
// Block = 8 waves (2 along M x 4 along N) -> 128x128 block tile.
// Wave = 4 Mtiles x 2 Ntiles (64 rows x 32 cols). Per 32-wide k-chunk:
// 16 A + 8 B b128 loads vs 24 WMMAs (1:1 issue ratio, minimal L0 pressure).
// Fragments load straight from row-major bf16 global memory: every WMMA A/B
// VGPR holds two K-contiguous bf16s, so loads coalesce into b128s.
__global__ void __launch_bounds__(256)
gemm_kernel(const unsigned short* __restrict__ Ah, const unsigned short* __restrict__ Al,
            const unsigned short* __restrict__ Bh, const unsigned short* __restrict__ Bl,
            float* __restrict__ Y, int Nout, int K) {
    int tid = threadIdx.x;
    int wave = tid >> 5, lane = tid & 31;
    int lane16 = lane & 15, half = lane >> 4;
    int wm = wave >> 2;                                // 0..1  (M direction)
    int wn = wave & 3;                                 // 0..3  (N direction)
    int mbase = (blockIdx.x * 8 + wm * 4) * 16;        // 4 m-tiles of 16 rows
    int colBase = blockIdx.y * 128 + wn * 32;          // 2 n-tiles of 16 cols

    const unsigned short* aph[4];
    const unsigned short* apl[4];
#pragma unroll
    for (int mt = 0; mt < 4; ++mt) {
        size_t r = (size_t)(mbase + mt * 16 + lane16) * K;
        aph[mt] = Ah + r;
        apl[mt] = Al + r;
    }
    const unsigned short* bph[2];
    const unsigned short* bpl[2];
#pragma unroll
    for (int nt = 0; nt < 2; ++nt) {
        // B 32x16 bf16 layout: K = half*8 + (j&3)*2 + (j>>2)*16 (+kb), N = lane16
        size_t r = (size_t)(colBase + nt * 16 + lane16) * K + half * 8;
        bph[nt] = Bh + r;
        bpl[nt] = Bl + r;
    }

    v8f acc[4][2] = {};
    int kchunks = K >> 5;
    for (int kc = 0; kc < kchunks; ++kc) {
        int kb = kc * 32;
        v16bf bh[2], bl[2];
#pragma unroll
        for (int nt = 0; nt < 2; ++nt) {
            v8u uh, ul;
#pragma unroll
            for (int j = 0; j < 8; ++j) {
                uh[j] = *(const unsigned int*)(bph[nt] + kb + ((j & 3) << 1) + ((j >> 2) << 4));
                ul[j] = *(const unsigned int*)(bpl[nt] + kb + ((j & 3) << 1) + ((j >> 2) << 4));
            }
            bh[nt] = __builtin_bit_cast(v16bf, uh);
            bl[nt] = __builtin_bit_cast(v16bf, ul);
        }
#pragma unroll
        for (int mt = 0; mt < 4; ++mt) {
            v8u uh, ul;
#pragma unroll
            for (int j = 0; j < 8; ++j) {
                // A 16x32 bf16 layout: K = half*8 + (j&3)*2 + (j>>2)*16 (+kb)
                int ao = kb + half * 8 + ((j & 3) << 1) + ((j >> 2) << 4);
                uh[j] = *(const unsigned int*)(aph[mt] + ao);
                ul[j] = *(const unsigned int*)(apl[mt] + ao);
            }
            v16bf ah = __builtin_bit_cast(v16bf, uh);
            v16bf al = __builtin_bit_cast(v16bf, ul);
#pragma unroll
            for (int nt = 0; nt < 2; ++nt) {
                // split-bf16 product: hi*hi + hi*lo + lo*hi (fp32-class accuracy)
                acc[mt][nt] = __builtin_amdgcn_wmma_f32_16x16x32_bf16(
                    false, ah, false, bh[nt], (short)0, acc[mt][nt], false, false);
                acc[mt][nt] = __builtin_amdgcn_wmma_f32_16x16x32_bf16(
                    false, ah, false, bl[nt], (short)0, acc[mt][nt], false, false);
                acc[mt][nt] = __builtin_amdgcn_wmma_f32_16x16x32_bf16(
                    false, al, false, bh[nt], (short)0, acc[mt][nt], false, false);
            }
        }
    }
#pragma unroll
    for (int mt = 0; mt < 4; ++mt) {
#pragma unroll
        for (int nt = 0; nt < 2; ++nt) {
            int col = colBase + nt * 16 + lane16;
#pragma unroll
            for (int r = 0; r < 8; ++r) {
                int row = mbase + mt * 16 + half * 8 + r;   // C/D f32 layout
                Y[(size_t)row * Nout + col] = acc[mt][nt][r];
            }
        }
    }
}

// ---- masked BN partial stats (deterministic: one partial per row-chunk) ----
__global__ void __launch_bounds__(256)
stats_kernel(const float* __restrict__ Y, int C, const int* __restrict__ plens,
             float* __restrict__ pSum, float* __restrict__ pSq) {
    int ch = blockIdx.x * 256 + threadIdx.x;
    if (ch >= C) return;
    int chunk = blockIdx.y;                 // NCHUNK chunks of 512 rows
    int rbase = chunk * (ROWS / NCHUNK);
    int b = rbase >> 11;
    int plen = plens[b];
    int nbase = rbase & (N_ - 1);
    const float* yp = Y + (size_t)rbase * C + ch;
    float s = 0.0f, q = 0.0f;
    for (int i = 0; i < ROWS / NCHUNK; ++i) {
        float v = yp[(size_t)i * C];
        float m = ((nbase + i) < plen) ? 1.0f : 0.0f;
        s += v * m;
        q += v * v * m;
    }
    pSum[(size_t)chunk * C + ch] = s;
    pSq[(size_t)chunk * C + ch] = q;
}

// ---- finalize BN: scale = g*rsqrt(var+eps), shift = b - mean*scale ---------
__global__ void __launch_bounds__(256)
bnfin_kernel(const float* __restrict__ pSum, const float* __restrict__ pSq,
             const float* __restrict__ g, const float* __restrict__ be,
             const int* __restrict__ plens, int C,
             float* __restrict__ scale, float* __restrict__ shift) {
    int ch = blockIdx.x * 256 + threadIdx.x;
    if (ch >= C) return;
    float cnt = 0.0f;
    for (int b = 0; b < B_; ++b) cnt += (float)plens[b];
    float s = 0.0f, q = 0.0f;
    for (int c = 0; c < NCHUNK; ++c) {
        s += pSum[(size_t)c * C + ch];
        q += pSq[(size_t)c * C + ch];
    }
    float mean = s / cnt;
    float var = q / cnt - mean * mean;
    float sc = g[ch] * rsqrtf(var + 1e-5f);
    scale[ch] = sc;
    shift[ch] = be[ch] - mean * sc;
}

// ---- BN apply + ReLU, output split bf16 hi/lo (input to next GEMM) ---------
__global__ void __launch_bounds__(256)
bnapply_split_kernel(const float* __restrict__ Y, const float* __restrict__ scale,
                     const float* __restrict__ shift, int C,
                     unsigned short* __restrict__ Hh, unsigned short* __restrict__ Hl) {
    size_t row = blockIdx.x;
    for (int c = threadIdx.x; c < C; c += 256) {
        float v = Y[row * C + c] * scale[c] + shift[c];
        v = v > 0.0f ? v : 0.0f;
        unsigned short h = f2bf(v);
        Hh[row * C + c] = h;
        Hl[row * C + c] = f2bf(v - bf2f(h));
    }
}

// ---- BN apply + ReLU, fp32 final output ------------------------------------
__global__ void __launch_bounds__(256)
bnapply_final_kernel(const float* __restrict__ Y, const float* __restrict__ scale,
                     const float* __restrict__ shift, int C, float* __restrict__ out) {
    size_t row = blockIdx.x;
    for (int c = threadIdx.x; c < C; c += 256) {
        float v = Y[row * C + c] * scale[c] + shift[c];
        out[row * C + c] = v > 0.0f ? v : 0.0f;
    }
}

// ---------------------------------------------------------------------------
extern "C" void kernel_launch(void* const* d_in, const int* in_sizes, int n_in,
                              void* d_out, int out_size, void* d_ws, size_t ws_size,
                              hipStream_t stream) {
    (void)in_sizes; (void)n_in; (void)out_size; (void)ws_size;
    const float* xyz1    = (const float*)d_in[0];
    const float* xyz2    = (const float*)d_in[1];
    const float* points1 = (const float*)d_in[2];
    const float* points2 = (const float*)d_in[3];
    const int*   plens   = (const int*)d_in[4];
    const int*   elens   = (const int*)d_in[5];
    /* d_in[6] = point_mask (bool) unused; recomputed from point_lens */
    const float* W0 = (const float*)d_in[7];
    const float* g0 = (const float*)d_in[8];
    const float* b0 = (const float*)d_in[9];
    const float* W1 = (const float*)d_in[10];
    const float* g1 = (const float*)d_in[11];
    const float* b1 = (const float*)d_in[12];
    float* out = (float*)d_out;

    // workspace carve (all 256B aligned)
    char* base = (char*)d_ws;
    size_t off = 0;
    auto alloc = [&](size_t bytes) -> void* {
        void* p = base + off;
        off = (off + bytes + 255) & ~(size_t)255;
        return p;
    };
    unsigned short* Xh  = (unsigned short*)alloc((size_t)ROWS * C0 * 2); // also H0h
    unsigned short* Xl  = (unsigned short*)alloc((size_t)ROWS * C0 * 2); // also H0l
    float*          Y   = (float*)alloc((size_t)ROWS * C0 * 4);          // Y0, then Y1
    unsigned short* W0h = (unsigned short*)alloc((size_t)C0 * C0 * 2);
    unsigned short* W0l = (unsigned short*)alloc((size_t)C0 * C0 * 2);
    unsigned short* W1h = (unsigned short*)alloc((size_t)C1 * C0 * 2);
    unsigned short* W1l = (unsigned short*)alloc((size_t)C1 * C0 * 2);
    int*   kIdx = (int*)alloc((size_t)ROWS * KNN * 4);
    float* kW   = (float*)alloc((size_t)ROWS * KNN * 4);
    float* ps0  = (float*)alloc((size_t)NCHUNK * C0 * 4);
    float* pq0  = (float*)alloc((size_t)NCHUNK * C0 * 4);
    float* ps1  = (float*)alloc((size_t)NCHUNK * C1 * 4);
    float* pq1  = (float*)alloc((size_t)NCHUNK * C1 * 4);
    float* sc0  = (float*)alloc(C0 * 4);
    float* sh0  = (float*)alloc(C0 * 4);
    float* sc1  = (float*)alloc(C1 * 4);
    float* sh1  = (float*)alloc(C1 * 4);

    // 1) bf16 hi/lo weight splits
    split_kernel<<<512, 256, 0, stream>>>(W0, W0h, W0l, C0 * C0);
    split_kernel<<<512, 256, 0, stream>>>(W1, W1h, W1l, C1 * C0);
    // 2) KNN + IDW weights
    knn_kernel<<<ROWS / 256, 256, 0, stream>>>(xyz1, xyz2, elens, kIdx, kW);
    // 3) X = [points1 | interp] as bf16 hi/lo
    buildx_kernel<<<ROWS, 256, 0, stream>>>(points1, points2, kIdx, kW, Xh, Xl);
    // 4) layer0 GEMM (M=32768, N=768, K=768) — split-bf16 WMMA
    gemm_kernel<<<dim3(ROWS / 128, C0 / 128), 256, 0, stream>>>(Xh, Xl, W0h, W0l, Y, C0, C0);
    // 5) masked BN 0 (deterministic two-stage reduction) + ReLU, re-split to bf16
    stats_kernel<<<dim3(3, NCHUNK), 256, 0, stream>>>(Y, C0, plens, ps0, pq0);
    bnfin_kernel<<<3, 256, 0, stream>>>(ps0, pq0, g0, b0, plens, C0, sc0, sh0);
    bnapply_split_kernel<<<ROWS, 256, 0, stream>>>(Y, sc0, sh0, C0, Xh, Xl);
    // 6) layer1 GEMM (M=32768, N=384, K=768)
    gemm_kernel<<<dim3(ROWS / 128, C1 / 128), 256, 0, stream>>>(Xh, Xl, W1h, W1l, Y, C1, C0);
    // 7) masked BN 1 + ReLU -> fp32 output
    stats_kernel<<<dim3(2, NCHUNK), 256, 0, stream>>>(Y, C1, plens, ps1, pq1);
    bnfin_kernel<<<2, 256, 0, stream>>>(ps1, pq1, g1, b1, plens, C1, sc1, sh1);
    bnapply_final_kernel<<<ROWS, 256, 0, stream>>>(Y, sc1, sh1, C1, out);
}
